// RNN_60979945669189
// MI455X (gfx1250) — matrix operand order repaired and checked by reading, not compiled, not run on hardware
//
#include <hip/hip_runtime.h>

#define B_    256
#define T_    512
#define IN_   512
#define HID_  1024
#define HALF_ 512
#define OUT_  64

typedef __attribute__((ext_vector_type(16))) __bf16 bf16x16;
typedef __attribute__((ext_vector_type(8)))  float  floatx8;
typedef int v4i __attribute__((vector_size(16)));

#define AS1 __attribute__((address_space(1)))
#define AS3 __attribute__((address_space(3)))

#if defined(__has_builtin)
#  if __has_builtin(__builtin_amdgcn_global_load_async_to_lds_b128)
#    define HAVE_ASYNC_LDS 1
#  endif
#endif
#ifndef HAVE_ASYNC_LDS
#  define HAVE_ASYNC_LDS 0
#endif

#if HAVE_ASYNC_LDS
#  if __has_builtin(__builtin_amdgcn_s_wait_asynccnt)
#    define ASYNC_WAIT() __builtin_amdgcn_s_wait_asynccnt(0)
#  else
#    define ASYNC_WAIT() asm volatile("s_wait_asynccnt 0x0" ::: "memory")
#  endif
#endif

// ---------------- WMMA helpers ----------------

__device__ __forceinline__ floatx8 wmma_bf16(bf16x16 a, bf16x16 b, floatx8 c) {
  return __builtin_amdgcn_wmma_f32_16x16x32_bf16(false, a, false, b, (short)0, c,
                                                 false, false);
}

// B fragment (32x16 bf16) from [N][K] row-major (weights are [out,in]).
__device__ __forceinline__ bf16x16 load_b_frag(const __bf16* __restrict__ src,
                                               int n_base, int k_base, int pitch) {
  int lane = threadIdx.x & 31;
  const __bf16* p = src + (size_t)(n_base + (lane & 15)) * pitch + k_base +
                    ((lane >> 4) << 3);
  union { uint4 u[2]; bf16x16 v; } cv;
  cv.u[0] = *(const uint4*)p;
  cv.u[1] = *(const uint4*)(p + 16);
  return cv.v;
}

// A fragment (16x32 bf16) from row-major [M][K] (global or LDS pointer).
__device__ __forceinline__ bf16x16 load_a_frag_bf16(const __bf16* src,
                                                    int m_base, int k_base, int pitch) {
  int lane = threadIdx.x & 31;
  const __bf16* p = src + (size_t)(m_base + (lane & 15)) * pitch + k_base +
                    ((lane >> 4) << 3);
  union { uint4 u[2]; bf16x16 v; } cv;
  cv.u[0] = *(const uint4*)p;
  cv.u[1] = *(const uint4*)(p + 16);
  return cv.v;
}

// A fragment from f32 packed x rows (pitch IN_), row index clipped like reference.
__device__ __forceinline__ bf16x16 load_a_frag_x(const float* __restrict__ x,
                                                 long xoff, int m_base, int k_base,
                                                 long nrows) {
  int lane = threadIdx.x & 31;
  long r = xoff + m_base + (lane & 15);
  if (r > nrows - 1) r = nrows - 1;
  const float* p = x + r * IN_ + k_base + ((lane >> 4) << 3);
  float4 a0 = *(const float4*)p;
  float4 a1 = *(const float4*)(p + 4);
  float4 a2 = *(const float4*)(p + 16);
  float4 a3 = *(const float4*)(p + 20);
  bf16x16 v;
  v[0]=(__bf16)a0.x;  v[1]=(__bf16)a0.y;  v[2]=(__bf16)a0.z;  v[3]=(__bf16)a0.w;
  v[4]=(__bf16)a1.x;  v[5]=(__bf16)a1.y;  v[6]=(__bf16)a1.z;  v[7]=(__bf16)a1.w;
  v[8]=(__bf16)a2.x;  v[9]=(__bf16)a2.y;  v[10]=(__bf16)a2.z; v[11]=(__bf16)a2.w;
  v[12]=(__bf16)a3.x; v[13]=(__bf16)a3.y; v[14]=(__bf16)a3.z; v[15]=(__bf16)a3.w;
  return v;
}

// A fragment from bf16 packed x rows (pitch IN_), clipped.
__device__ __forceinline__ bf16x16 load_a_frag_xbf(const __bf16* __restrict__ src,
                                                   long xoff, int m_base, int k_base,
                                                   long nrows) {
  int lane = threadIdx.x & 31;
  long r = xoff + m_base + (lane & 15);
  if (r > nrows - 1) r = nrows - 1;
  const __bf16* p = src + r * IN_ + k_base + ((lane >> 4) << 3);
  union { uint4 u[2]; bf16x16 v; } cv;
  cv.u[0] = *(const uint4*)p;
  cv.u[1] = *(const uint4*)(p + 16);
  return cv.v;
}

// ---------------- setup kernels ----------------

__global__ void cvt_f32_bf16(const float* __restrict__ s, __bf16* __restrict__ d, int n) {
  int i = blockIdx.x * blockDim.x + threadIdx.x;
  int stride = gridDim.x * blockDim.x;
  for (; i < n; i += stride) d[i] = (__bf16)s[i];
}

__global__ void bias_sum(const float* __restrict__ a, const float* __restrict__ b,
                         float* __restrict__ d, int n) {
  int i = blockIdx.x * blockDim.x + threadIdx.x;
  if (i < n) d[i] = a[i] + b[i];
}

// ---------------- hoisted input projection: Z = X @ U^T (fully parallel) ----------

__global__ __launch_bounds__(256) void xproj(const float* __restrict__ x,
                                             const __bf16* __restrict__ Ubf,
                                             float* __restrict__ Z, long nrows) {
  long ntr = (nrows + 15) >> 4;
  long task = (long)blockIdx.x * (blockDim.x >> 5) + (threadIdx.x >> 5);
  if (task >= ntr * 32) return;            // uniform within wave
  long mb = (task >> 5) * 16;
  int  nb0 = (int)(task & 31) * 32;
  floatx8 acc0 = {0.f,0.f,0.f,0.f,0.f,0.f,0.f,0.f};
  floatx8 acc1 = {0.f,0.f,0.f,0.f,0.f,0.f,0.f,0.f};
#pragma unroll 4
  for (int kb = 0; kb < IN_; kb += 32) {
    bf16x16 a = load_a_frag_x(x, mb, 0, kb, nrows);
    acc0 = wmma_bf16(a, load_b_frag(Ubf, nb0,      kb, IN_), acc0);
    acc1 = wmma_bf16(a, load_b_frag(Ubf, nb0 + 16, kb, IN_), acc1);
  }
  int lane = threadIdx.x & 31;
  int col0 = nb0 + (lane & 15);
  long rb  = mb + ((lane >> 4) << 3);
#pragma unroll
  for (int r = 0; r < 8; ++r) {
    long row = rb + r;
    if (row < nrows) {
      Z[row * HID_ + col0]      = acc0[r];
      Z[row * HID_ + col0 + 16] = acc1[r];
    }
  }
}

// ---------------- recurrence (Z-hoisted): h = tanh(Z_t + h @ W^T + b) -----------
// Block g owns rows [16g,16g+16). H lives in LDS (32 KB), updated IN PLACE each
// step between two barriers -> frozen rows keep their final value for free.
// Z slice for the next step is staged into LDS with CDNA5 async global->LDS
// loads (ASYNCcnt) when available; else direct loads + prefetch.

#if HAVE_ASYNC_LDS
__device__ __forceinline__ void stage_z_async(const float* __restrict__ Z, long row0,
                                              long nrows, float* dst) {
  // 16 rows x 1024 f32 = 4096 x b128 ; 4 async issues per thread (1024 thr)
  for (int idx = threadIdx.x; idx < (16 * HID_) / 4; idx += blockDim.x) {
    long rg = row0 + (idx >> 8);
    if (rg > nrows - 1) rg = nrows - 1;
    float* g = const_cast<float*>(Z) + rg * HID_ + (size_t)(idx & 255) * 4;
    float* l = dst + (size_t)idx * 4;
    __builtin_amdgcn_global_load_async_to_lds_b128(
        (AS1 v4i*)(AS1 void*)g, (AS3 v4i*)(AS3 void*)l, 0, 0);
  }
}
#endif

__global__ __launch_bounds__(1024) void rnn_recur_z(
    const int* __restrict__ bsz, const float* __restrict__ Z,
    const __bf16* __restrict__ Wbf, const float* __restrict__ bias_hw,
    __bf16* __restrict__ Hout, long nrows)
{
  __shared__ __align__(16) __bf16 Hs[16 * HID_];
#if HAVE_ASYNC_LDS
  __shared__ __align__(16) float Zs[2][16 * HID_];
#endif
  const int mb   = blockIdx.x * 16;
  const int nb0  = (threadIdx.x >> 5) * 32;
  const int lane = threadIdx.x & 31;

  for (int i = threadIdx.x; i < 16 * HID_ / 8; i += blockDim.x)
    ((uint4*)Hs)[i] = make_uint4(0, 0, 0, 0);
#if HAVE_ASYNC_LDS
  stage_z_async(Z, (long)mb, nrows, Zs[0]);   // Z slice for t = 0
  ASYNC_WAIT();
#endif
  __syncthreads();

  long xoff = 0;
  for (int t = 0; t < T_; ++t) {
    int bs = bsz[t];
    if (bs <= mb) break;                 // uniform; Hs already holds finals
    long xoff_n = xoff + bs;
#if HAVE_ASYNC_LDS
    if (t + 1 < T_) stage_z_async(Z, xoff_n + mb, nrows, Zs[(t + 1) & 1]);
#endif

    floatx8 acc0 = {0.f,0.f,0.f,0.f,0.f,0.f,0.f,0.f};
    floatx8 acc1 = {0.f,0.f,0.f,0.f,0.f,0.f,0.f,0.f};
#pragma unroll 4
    for (int kb = 0; kb < HID_; kb += 32) {
      bf16x16 a = load_a_frag_bf16(Hs, 0, kb, HID_);         // ds_load_b128
      acc0 = wmma_bf16(a, load_b_frag(Wbf, nb0,      kb, HID_), acc0);
      acc1 = wmma_bf16(a, load_b_frag(Wbf, nb0 + 16, kb, HID_), acc1);
    }
    __syncthreads();                     // all Hs reads done before in-place write

    int col0 = nb0 + (lane & 15);
    int rl   = (lane >> 4) << 3;
    float b0 = bias_hw[col0];
    float b1 = bias_hw[col0 + 16];
#pragma unroll
    for (int r = 0; r < 8; ++r) {
      int row = mb + rl + r;
      if (row < bs) {
#if HAVE_ASYNC_LDS
        const float* zp = Zs[t & 1] + (rl + r) * HID_;
#else
        const float* zp = Z + (size_t)(xoff + row) * HID_;
#endif
        Hs[(rl + r) * HID_ + col0]      = (__bf16)tanhf(acc0[r] + b0 + zp[col0]);
        Hs[(rl + r) * HID_ + col0 + 16] = (__bf16)tanhf(acc1[r] + b1 + zp[col0 + 16]);
      }
    }
#if HAVE_ASYNC_LDS
    ASYNC_WAIT();                        // next-step Z staged before publishing
#else
    __builtin_prefetch(Z + (size_t)(xoff_n + mb + rl) * HID_ + col0, 0, 0);
#endif
    xoff = xoff_n;
    __syncthreads();                     // publish Hs (and Zs) for next step
  }
  __syncthreads();
  for (int i = threadIdx.x; i < 16 * HID_ / 8; i += blockDim.x)
    ((uint4*)(Hout + (size_t)mb * HID_))[i] = ((const uint4*)Hs)[i];
}

// ---------------- recurrence fallback (U GEMM in-loop), MODE 0: f32 x, 1: bf16 x --

template <int MODE>
__global__ __launch_bounds__(1024) void rnn_recur_fb(
    const float* __restrict__ x, const __bf16* __restrict__ Xbf,
    const int* __restrict__ bsz,
    const __bf16* __restrict__ Ubf, const __bf16* __restrict__ Wbf,
    const float* __restrict__ bias_hw, __bf16* __restrict__ Hout, long nrows)
{
  __shared__ __align__(16) __bf16 Hs[16 * HID_];
  const int mb   = blockIdx.x * 16;
  const int nb0  = (threadIdx.x >> 5) * 32;
  const int lane = threadIdx.x & 31;

  for (int i = threadIdx.x; i < 16 * HID_ / 8; i += blockDim.x)
    ((uint4*)Hs)[i] = make_uint4(0, 0, 0, 0);
  __syncthreads();

  long xoff = 0;
  for (int t = 0; t < T_; ++t) {
    int bs = bsz[t];
    if (bs <= mb) break;

    floatx8 acc0 = {0.f,0.f,0.f,0.f,0.f,0.f,0.f,0.f};
    floatx8 acc1 = {0.f,0.f,0.f,0.f,0.f,0.f,0.f,0.f};
#pragma unroll 4
    for (int kb = 0; kb < IN_; kb += 32) {
      bf16x16 a = (MODE == 0) ? load_a_frag_x(x, xoff, mb, kb, nrows)
                              : load_a_frag_xbf(Xbf, xoff, mb, kb, nrows);
      acc0 = wmma_bf16(a, load_b_frag(Ubf, nb0,      kb, IN_), acc0);
      acc1 = wmma_bf16(a, load_b_frag(Ubf, nb0 + 16, kb, IN_), acc1);
    }
#pragma unroll 4
    for (int kb = 0; kb < HID_; kb += 32) {
      bf16x16 a = load_a_frag_bf16(Hs, 0, kb, HID_);
      acc0 = wmma_bf16(a, load_b_frag(Wbf, nb0,      kb, HID_), acc0);
      acc1 = wmma_bf16(a, load_b_frag(Wbf, nb0 + 16, kb, HID_), acc1);
    }
    __syncthreads();

    int col0 = nb0 + (lane & 15);
    int rl   = (lane >> 4) << 3;
    float b0 = bias_hw[col0];
    float b1 = bias_hw[col0 + 16];
#pragma unroll
    for (int r = 0; r < 8; ++r) {
      int row = mb + rl + r;
      if (row < bs) {
        Hs[(rl + r) * HID_ + col0]      = (__bf16)tanhf(acc0[r] + b0);
        Hs[(rl + r) * HID_ + col0 + 16] = (__bf16)tanhf(acc1[r] + b1);
      }
    }
    xoff += bs;
    __syncthreads();
  }
  __syncthreads();
  for (int i = threadIdx.x; i < 16 * HID_ / 8; i += blockDim.x)
    ((uint4*)(Hout + (size_t)mb * HID_))[i] = ((const uint4*)Hs)[i];
}

// ---------------- epilogue GEMMs ----------------

__global__ __launch_bounds__(1024) void out_gate(
    const __bf16* __restrict__ H, const __bf16* __restrict__ Vbf,
    const float* __restrict__ v_b, __bf16* __restrict__ Obf)
{
  int mb = blockIdx.x * 16;
  int nb = (threadIdx.x >> 5) * 16;
  floatx8 acc = {0.f,0.f,0.f,0.f,0.f,0.f,0.f,0.f};
#pragma unroll 4
  for (int kb = 0; kb < HID_; kb += 32)
    acc = wmma_bf16(load_a_frag_bf16(H, mb, kb, HID_),
                    load_b_frag(Vbf, nb, kb, HID_), acc);
  int lane = threadIdx.x & 31;
  int col  = nb + (lane & 15);
  int rb   = mb + ((lane >> 4) << 3);
  float bv = v_b[col];
#pragma unroll
  for (int r = 0; r < 8; ++r) {
    float v = acc[r] + bv;
    v = 1.0f / (1.0f + __expf(-v));
    Obf[(size_t)(rb + r) * HALF_ + col] = (__bf16)v;
  }
}

__global__ __launch_bounds__(128) void y_kernel(
    const __bf16* __restrict__ Obf, const __bf16* __restrict__ Cbf,
    const float* __restrict__ c_b, const int* __restrict__ sidx,
    float* __restrict__ y)
{
  int mb = blockIdx.x * 16;
  int nb = (threadIdx.x >> 5) * 16;
  floatx8 acc = {0.f,0.f,0.f,0.f,0.f,0.f,0.f,0.f};
#pragma unroll 4
  for (int kb = 0; kb < HALF_; kb += 32)
    acc = wmma_bf16(load_a_frag_bf16(Obf, mb, kb, HALF_),
                    load_b_frag(Cbf, nb, kb, HALF_), acc);
  int lane = threadIdx.x & 31;
  int col  = nb + (lane & 15);
  int rb   = mb + ((lane >> 4) << 3);
  float bc = c_b[col];
#pragma unroll
  for (int r = 0; r < 8; ++r) {
    int orow = sidx[rb + r];
    y[(size_t)orow * OUT_ + col] = acc[r] + bc;
  }
}

__global__ void h_scatter(const __bf16* __restrict__ H, const int* __restrict__ sidx,
                          float* __restrict__ outh)
{
  int i = blockIdx.x * blockDim.x + threadIdx.x;
  int stride = gridDim.x * blockDim.x;
  for (; i < B_ * HID_; i += stride) {
    int r = i >> 10, c = i & (HID_ - 1);
    outh[(size_t)sidx[r] * HID_ + c] = (float)H[i];
  }
}

// ---------------- launch ----------------

extern "C" void kernel_launch(void* const* d_in, const int* in_sizes, int n_in,
                              void* d_out, int out_size, void* d_ws, size_t ws_size,
                              hipStream_t stream) {
  const float* x    = (const float*)d_in[0];
  const int*   bsz  = (const int*)  d_in[1];
  const int*   sidx = (const int*)  d_in[2];
  const float* u_w  = (const float*)d_in[3];
  const float* u_b  = (const float*)d_in[4];
  const float* w_w  = (const float*)d_in[5];
  const float* w_b  = (const float*)d_in[6];
  const float* v_w  = (const float*)d_in[7];
  const float* v_b  = (const float*)d_in[8];
  const float* c_w  = (const float*)d_in[9];
  const float* c_b  = (const float*)d_in[10];
  float* out = (float*)d_out;
  long nrows = (long)in_sizes[0] / IN_;

  char* ws = (char*)d_ws;
  __bf16* Ubf = (__bf16*)ws;   ws += (size_t)HID_  * IN_   * 2;
  __bf16* Wbf = (__bf16*)ws;   ws += (size_t)HID_  * HID_  * 2;
  __bf16* Vbf = (__bf16*)ws;   ws += (size_t)HALF_ * HID_  * 2;
  __bf16* Cbf = (__bf16*)ws;   ws += (size_t)OUT_  * HALF_ * 2;
  float* bias_hw = (float*)ws; ws += (size_t)HID_ * 4;
  __bf16* Hbf = (__bf16*)ws;   ws += (size_t)B_ * HID_  * 2;
  __bf16* Obf = (__bf16*)ws;   ws += (size_t)B_ * HALF_ * 2;
  size_t base_used = (size_t)(ws - (char*)d_ws);

  cvt_f32_bf16<<<512, 256, 0, stream>>>(u_w, Ubf, HID_ * IN_);
  cvt_f32_bf16<<<512, 256, 0, stream>>>(w_w, Wbf, HID_ * HID_);
  cvt_f32_bf16<<<512, 256, 0, stream>>>(v_w, Vbf, HALF_ * HID_);
  cvt_f32_bf16<<<64,  256, 0, stream>>>(c_w, Cbf, OUT_ * HALF_);
  bias_sum<<<4, 256, 0, stream>>>(u_b, w_b, bias_hw, HID_);

  size_t z_bytes   = (size_t)nrows * HID_ * sizeof(float);
  size_t xbf_bytes = (size_t)nrows * IN_ * sizeof(__bf16);

  if (ws_size >= base_used + z_bytes) {
    float* Z = (float*)ws;
    long ntasks = ((nrows + 15) >> 4) * 32;
    int nblocks = (int)((ntasks + 7) / 8);
    xproj<<<nblocks, 256, 0, stream>>>(x, Ubf, Z, nrows);
    rnn_recur_z<<<16, 1024, 0, stream>>>(bsz, Z, Wbf, bias_hw, Hbf, nrows);
  } else if (ws_size >= base_used + xbf_bytes) {
    __bf16* Xbf = (__bf16*)ws;
    cvt_f32_bf16<<<4096, 256, 0, stream>>>(x, Xbf, (int)(nrows * IN_));
    rnn_recur_fb<1><<<16, 1024, 0, stream>>>(x, Xbf, bsz, Ubf, Wbf, bias_hw, Hbf, nrows);
  } else {
    rnn_recur_fb<0><<<16, 1024, 0, stream>>>(x, (const __bf16*)nullptr, bsz, Ubf, Wbf,
                                             bias_hw, Hbf, nrows);
  }

  out_gate<<<16, 1024, 0, stream>>>(Hbf, Vbf, v_b, Obf);
  y_kernel<<<16, 128, 0, stream>>>(Obf, Cbf, c_b, sidx, out);
  h_scatter<<<512, 256, 0, stream>>>(Hbf, sidx, out + (size_t)B_ * OUT_);
}
